// Structure2VecLayer_40922448396570
// MI455X (gfx1250) — matrix operand order repaired and checked by reading, not compiled, not run on hardware
//
#include <hip/hip_runtime.h>
#include <hip/hip_bf16.h>

typedef __attribute__((ext_vector_type(16))) _Float16 v16h;
typedef __attribute__((ext_vector_type(8)))  _Float16 v8h;
typedef __attribute__((ext_vector_type(8)))  float    v8f;

#define HID 128
#define LDS_STRIDE 136   // 128 + 8 halves padding; 136*2=272 bytes = 16*17 (16B aligned rows)
#define MREP 2           // M tiles (of 16 rows) per block

__device__ __forceinline__ void atomAddF(float* p, float v) {
    unsafeAtomicAdd(p, v);   // global_atomic_add_f32
}

// byte offset of a __shared__ object within the workgroup LDS allocation
__device__ __forceinline__ unsigned lds_offset(const void* p) {
    return (unsigned)(size_t)(const __attribute__((address_space(3))) void*)p;
}

// ---------------------------------------------------------------- zero scratch
__global__ void zero_kernel(float* __restrict__ p, int count) {
    int i = blockIdx.x * blockDim.x + threadIdx.x;
    int stride = gridDim.x * blockDim.x;
    for (; i < count; i += stride) p[i] = 0.0f;
}

// ------------------------------------------------------ edge linear + scatter
// blockDim = 128 (thread t = hidden channel). 16 edges per block.
// h2[dst] += Wb @ ef + bb ; h1[dst] += features[src]
__global__ __launch_bounds__(128) void edge_kernel(
    const float* __restrict__ feats, const float* __restrict__ ef,
    const int*   __restrict__ src,   const int*   __restrict__ dst,
    const float* __restrict__ Wb,    const float* __restrict__ bb,
    float* __restrict__ h1, float* __restrict__ h2, int nEdges)
{
    __shared__ float Wbl[HID * 16];
    __shared__ float bbl[HID];
    __shared__ float efl[16 * 16];
    __shared__ int   sj[16], dj[16];

    const int t = threadIdx.x;
    for (int i = t; i < HID * 16; i += 128) Wbl[i] = Wb[i];
    bbl[t] = bb[t];

    const int e0 = blockIdx.x * 16;
    for (int i = t; i < 256; i += 128) {
        int e = e0 + (i >> 4);
        efl[i] = (e < nEdges) ? ef[(unsigned)e * 16u + (i & 15)] : 0.0f;
    }
    if (t < 16) {
        int e = e0 + t;
        sj[t] = (e < nEdges) ? src[e] : 0;
        dj[t] = (e < nEdges) ? dst[e] : 0;
    }
    __syncthreads();

    float wrow[16];
#pragma unroll
    for (int b = 0; b < 16; ++b) wrow[b] = Wbl[t * 16 + b];
    const float bias = bbl[t];

    for (int j = 0; j < 16; ++j) {
        int e = e0 + j;
        if (e >= nEdges) break;
        const unsigned s = (unsigned)sj[j], d = (unsigned)dj[j];
        float acc = bias;
#pragma unroll
        for (int b = 0; b < 16; ++b) acc += wrow[b] * efl[j * 16 + b];
        atomAddF(&h2[(d << 7) + t], acc);
        atomAddF(&h1[(d << 7) + t], feats[(s << 7) + t]);
    }
}

// ----------------------------------------- T = A @ W^T + bias + add, + BN stats
// One block = 32 rows (2 M-tiles) x 128 cols; 256 threads = 8 waves, one 16-col
// WMMA tile per wave. W & A staged f16 in LDS; B fragments register-resident and
// reused across M-tiles; `add` tile streamed via global_load_async_to_lds_b128
// overlapped with the WMMA K-loops.
__global__ __launch_bounds__(256) void gemm_bn_kernel(
    const float* __restrict__ A,    const float* __restrict__ W,
    const float* __restrict__ bias, const float* __restrict__ add,
    float* __restrict__ T, float* __restrict__ colsum, float* __restrict__ colsumsq,
    int nRows)
{
    __shared__ __align__(16) _Float16 Wl[HID * LDS_STRIDE];        // 34816 B
    __shared__ __align__(16) _Float16 Al[16 * MREP * LDS_STRIDE];  //  8704 B
    __shared__ __align__(16) float    AddL[16 * MREP * HID];       // 16384 B

    const int t = threadIdx.x;
    const int m0 = blockIdx.x * (16 * MREP);
    const int lastRow = nRows - 1;

    // stage W (128x128 f32 -> f16), row-major [out][k]
    for (int i = t; i < HID * HID; i += 256)
        Wl[(i >> 7) * LDS_STRIDE + (i & 127)] = (_Float16)W[i];

    // stage A tile (32x128 f32 -> f16), clamp OOB rows to a valid row
    for (int i = t; i < 16 * MREP * HID; i += 256) {
        int r = i >> 7, k = i & 127;
        int m = m0 + r; m = m > lastRow ? lastRow : m;
        Al[r * LDS_STRIDE + k] = (_Float16)A[(unsigned)m * HID + k];
    }
    __syncthreads();

    // async-stage the `add` tile (32x128 f32, 16B chunks) -> overlaps K-loop
    {
        const unsigned ldsBase = lds_offset(AddL);
#pragma unroll
        for (int i = 0; i < (16 * MREP * HID) / 4 / 256; ++i) {     // 4 chunks/thread
            int idx = t + i * 256;                                  // 16B chunk id
            int r = idx >> 5, c = idx & 31;
            int m = m0 + r; m = m > lastRow ? lastRow : m;
            unsigned voff   = (unsigned)m * (HID * 4u) + (unsigned)c * 16u;
            unsigned ldsoff = ldsBase + (unsigned)(r * HID * 4 + c * 16);
            asm volatile("global_load_async_to_lds_b128 %0, %1, %2"
                         :: "v"(ldsoff), "v"(voff), "s"(add) : "memory");
        }
    }

    const int lane = t & 31;
    const int hi   = lane >> 4;      // half-wave select
    const int nc   = lane & 15;      // column within tile
    const int row  = lane & 15;      // A row within tile
    const int n0   = (t >> 5) * 16;  // wave's 16-column slice
    const int n    = n0 + nc;
    const float bn = bias[n];

    // B fragments: load once, reuse across M-tiles.
    // Lane layout (32x16 f16): lanes 0-15 hold K=ks+0..7,ks+16..23 of column n,
    // lanes 16-31 hold K=ks+8..15,ks+24..31.
    v16h bfrag[4];
#pragma unroll
    for (int kk = 0; kk < 4; ++kk) {
        const _Float16* bp = &Wl[n * LDS_STRIDE + kk * 32 + hi * 8];
        v8h blo = *(const v8h*)bp;
        v8h bhi = *(const v8h*)(bp + 16);
        bfrag[kk] = __builtin_shufflevector(blo, bhi, 0,1,2,3,4,5,6,7,8,9,10,11,12,13,14,15);
    }

    // K-loops for both M-tiles (A fragment: lanes 0-15 hold K=ks+0..7,ks+16..23;
    // lanes 16-31 hold K=ks+8..15,ks+24..31).
    v8f acc[MREP];
#pragma unroll
    for (int mt = 0; mt < MREP; ++mt) {
        acc[mt] = (v8f){};
#pragma unroll
        for (int kk = 0; kk < 4; ++kk) {
            const _Float16* ap = &Al[(mt * 16 + row) * LDS_STRIDE + kk * 32 + hi * 8];
            v8h alo = *(const v8h*)ap;
            v8h ahi = *(const v8h*)(ap + 16);
            v16h a = __builtin_shufflevector(alo, ahi, 0,1,2,3,4,5,6,7,8,9,10,11,12,13,14,15);
            acc[mt] = __builtin_amdgcn_wmma_f32_16x16x32_f16(
                false, a, false, bfrag[kk], (short)0, acc[mt], false, false);
        }
    }

    // wait for async `add` tile, then barrier so all producers are done
    asm volatile("s_wait_asynccnt 0" ::: "memory");
    __syncthreads();

    // Epilogue. D layout: VGPR r holds row (r + 8*hi), col nc.
    float ps = 0.0f, pss = 0.0f;
    if (m0 + 16 * MREP <= nRows) {       // fast path: full tile, 32-bit offsets
#pragma unroll
        for (int mt = 0; mt < MREP; ++mt) {
#pragma unroll
            for (int r = 0; r < 8; ++r) {
                int rl = mt * 16 + r + hi * 8;
                float v = acc[mt][r] + bn + AddL[rl * HID + n];
                T[(unsigned)(m0 + rl) * HID + n] = v;
                ps += v; pss += v * v;
            }
        }
    } else {
#pragma unroll
        for (int mt = 0; mt < MREP; ++mt) {
#pragma unroll
            for (int r = 0; r < 8; ++r) {
                int rl = mt * 16 + r + hi * 8;
                int m = m0 + rl;
                if (m < nRows) {
                    float v = acc[mt][r] + bn + AddL[rl * HID + n];
                    T[(unsigned)m * HID + n] = v;
                    ps += v; pss += v * v;
                }
            }
        }
    }

    // lanes l and l+16 share column n: pairwise reduce, single atomic per column
    ps  += __shfl_xor(ps, 16);
    pss += __shfl_xor(pss, 16);
    if (hi == 0) {
        atomAddF(&colsum[n], ps);
        atomAddF(&colsumsq[n], pss);
    }
}

// ------------------------------------------------------------- BN stat folding
__global__ void bnstats_kernel(const float* __restrict__ s, const float* __restrict__ sq,
                               const float* __restrict__ gamma, const float* __restrict__ beta,
                               float* __restrict__ scale, float* __restrict__ shift,
                               float inv_n)
{
    int n = threadIdx.x;  // 128 threads
    float mean = s[n] * inv_n;
    float var  = sq[n] * inv_n - mean * mean;
    float a    = gamma[n] * rsqrtf(var + 1e-5f);
    scale[n] = a;
    shift[n] = beta[n] - mean * a;
}

// -------------------------------------------------------------- BN apply + ReLU
__global__ void bnrelu_kernel(const float* __restrict__ T,
                              const float* __restrict__ scale, const float* __restrict__ shift,
                              float* __restrict__ out, int total)
{
    int i = blockIdx.x * blockDim.x + threadIdx.x;
    int stride = gridDim.x * blockDim.x;
    for (; i < total; i += stride) {
        int n = i & (HID - 1);
        float v = T[i] * scale[n] + shift[n];
        out[i] = v > 0.0f ? v : 0.0f;
    }
}

extern "C" void kernel_launch(void* const* d_in, const int* in_sizes, int n_in,
                              void* d_out, int out_size, void* d_ws, size_t ws_size,
                              hipStream_t stream) {
    const float* features = (const float*)d_in[0];
    const float* edgef    = (const float*)d_in[1];
    const int*   src      = (const int*)d_in[2];
    const int*   dst      = (const int*)d_in[3];
    const float* Wb       = (const float*)d_in[4];
    const float* bb       = (const float*)d_in[5];
    const float* W1       = (const float*)d_in[6];
    const float* b1       = (const float*)d_in[7];
    const float* W2       = (const float*)d_in[8];
    const float* b2       = (const float*)d_in[9];
    const float* gamma1   = (const float*)d_in[10];
    const float* beta1    = (const float*)d_in[11];
    const float* gamma2   = (const float*)d_in[12];
    const float* beta2    = (const float*)d_in[13];
    float* out = (float*)d_out;

    const int n = in_sizes[0] / HID;   // nodes
    const int e = in_sizes[2];         // edges
    const int total = n * HID;

    // workspace layout: h1 acc | h2 acc | stats (8 x 128)
    float* ws = (float*)d_ws;
    float* h1 = ws;
    float* h2 = ws + total;
    float* st = h2 + total;
    float *s1 = st, *sq1 = st + 128, *sc1 = st + 256, *sh1 = st + 384;
    float *s2 = st + 512, *sq2 = st + 640, *sc2 = st + 768, *sh2 = st + 896;

    // 1. zero accumulators (h1, h2, stat sums)
    zero_kernel<<<1024, 256, 0, stream>>>(ws, 2 * total + 1024);

    // 2. edge linear + gather/scatter-add mailboxes
    edge_kernel<<<(e + 15) / 16, 128, 0, stream>>>(
        features, edgef, src, dst, Wb, bb, h1, h2, e);

    const int mtiles = (n + 16 * MREP - 1) / (16 * MREP);

    // 3. layer 1: t1 = h1 @ W1^T + b1 + h2 (into d_out), + BN stats
    gemm_bn_kernel<<<mtiles, 256, 0, stream>>>(h1, W1, b1, h2, out, s1, sq1, n);
    bnstats_kernel<<<1, 128, 0, stream>>>(s1, sq1, gamma1, beta1, sc1, sh1, 1.0f / n);
    bnrelu_kernel<<<2048, 256, 0, stream>>>(out, sc1, sh1, h1, total);   // hA -> h1 buf

    // 4. layer 2: t2 = hA @ W2^T + b2 + features (into h2 buf)
    gemm_bn_kernel<<<mtiles, 256, 0, stream>>>(h1, W2, b2, features, h2, s2, sq2, n);
    bnstats_kernel<<<1, 128, 0, stream>>>(s2, sq2, gamma2, beta2, sc2, sh2, 1.0f / n);
    bnrelu_kernel<<<2048, 256, 0, stream>>>(h2, sc2, sh2, out, total);
}
